// MultiLevel_DGCN_LSTM_49443663511667
// MI455X (gfx1250) — matrix operand order repaired and checked by reading, not compiled, not run on hardware
//
#include <hip/hip_runtime.h>
#include <hip/hip_bf16.h>
#include <stdint.h>

// Problem constants (match reference)
#define NN 10000
#define TT 8
#define CC 16
#define OO 64
#define KK 16
#define LL 3

#define TILE 64       // nodes per block in LSTM kernel
#define NTHREADS 128  // 4 waves; each wave owns 16 rows

typedef __attribute__((ext_vector_type(16))) __bf16 v16bf;
typedef __attribute__((ext_vector_type(8)))  float  v8f;

__device__ __forceinline__ unsigned short f2bf(float f) {
  unsigned int u = __float_as_uint(f);
  u += 0x7FFFu + ((u >> 16) & 1u);   // round-to-nearest-even
  return (unsigned short)(u >> 16);
}

// Deterministic float atomic-max via monotone integer encodings.
// Valid given destination initialized to -inf.
__device__ __forceinline__ void atomicMaxF(float* addr, float v) {
  if (v >= 0.0f) atomicMax((int*)addr, __float_as_int(v));
  else           atomicMin((unsigned int*)addr, __float_as_uint(v));
}

__device__ __forceinline__ v8f wmma_bf16(v16bf a, v16bf b, v8f c) {
  // v_wmma_f32_16x16x32_bf16: D = A(16x32) x B(32x16) + C(16x16 f32)
  return __builtin_amdgcn_wmma_f32_16x16x32_bf16(
      false, a, false, b, (short)0, c, false, false);
}

union Frag { uint4 u[2]; v16bf v; };

// A-matrix 16x32 bf16 fragment from LDS row-major [rows][64] at (m0, k0).
// ISA layout: lane<16 -> M=lane, K in {kb..kb+7, kb+16..kb+23} with kb=k0;
// lane>=16 -> kb=k0+8.
__device__ __forceinline__ v16bf load_afrag(const unsigned short* base,
                                            int m0, int k0, int lane) {
  const int row = m0 + (lane & 15);
  const int kb  = k0 + ((lane >> 4) << 3);
  const unsigned short* p = base + row * OO + kb;
  Frag f;
  f.u[0] = *(const uint4*)(p);        // K = kb .. kb+7
  f.u[1] = *(const uint4*)(p + 16);   // K = kb+16 .. kb+23
  return f.v;
}

// B-matrix 32x16 bf16 fragment, B[k][j] = W[j][k], W row-major [256][64].
// ISA layout: lane<16 -> N=lane, K in {kb..kb+7, kb+16..kb+23} with kb=k0;
// lane>=16 -> kb=k0+8.
__device__ __forceinline__ v16bf load_bfrag(const unsigned short* W,
                                            int j0, int k0, int lane) {
  const int col = j0 + (lane & 15);
  const int kb  = k0 + ((lane >> 4) << 3);
  const unsigned short* p = W + col * OO + kb;
  Frag f;
  f.u[0] = *(const uint4*)(p);        // K = kb .. kb+7
  f.u[1] = *(const uint4*)(p + 16);   // K = kb+16 .. kb+23
  return f.v;
}

// ---------------- Kernel 0: init output to -inf ----------------
__global__ void init_out_kernel(float* __restrict__ out, int n) {
  int i = blockIdx.x * blockDim.x + threadIdx.x;
  if (i < n) out[i] = __uint_as_float(0xFF800000u); // -inf
}

// ---------------- Kernel 1: P[l,t,n,o] = X[n,t,:] . conv_w[l,o,:] ---------
__global__ __launch_bounds__(256)
void conv_p_kernel(const float* __restrict__ X,
                   const float* __restrict__ conv_w,
                   float* __restrict__ P) {
  __shared__ float w[OO * CC];
  const int l = blockIdx.z, t = blockIdx.y;
  const int tileBase = blockIdx.x * 64;
  for (int i = threadIdx.x; i < OO * CC; i += blockDim.x)
    w[i] = conv_w[l * OO * CC + i];
  __syncthreads();
  const int node = tileBase + (threadIdx.x >> 2);
  const int og   = (threadIdx.x & 3) * 16;
  if (node >= NN) return;
  float x[CC];
  const float* xp = X + ((size_t)node * TT + t) * CC;
#pragma unroll
  for (int c = 0; c < CC; ++c) x[c] = xp[c];
  float* pp = P + (((size_t)l * TT + t) * NN + node) * OO + og;
#pragma unroll
  for (int o = 0; o < 16; ++o) {
    float acc = 0.f;
#pragma unroll
    for (int c = 0; c < CC; ++c) acc += x[c] * w[(og + o) * CC + c];
    pp[o] = acc;
  }
}

// ---------------- Kernel 2: gather-max + LSTM + level-max aggregate -------
__global__ __launch_bounds__(NTHREADS)
void dgcn_lstm_kernel(const float* __restrict__ P,
                      const int*   __restrict__ As,
                      const float* __restrict__ conv_b,
                      const float* __restrict__ W_ih,
                      const float* __restrict__ W_hh,
                      const float* __restrict__ b_ih,
                      const float* __restrict__ b_hh,
                      float* __restrict__ out) {
  __shared__ __align__(16) unsigned short sWih[4 * OO * OO];  // 32 KB bf16
  __shared__ __align__(16) unsigned short sWhh[4 * OO * OO];  // 32 KB bf16
  __shared__ __align__(16) unsigned short sFeat[TILE * OO];   //  8 KB bf16
  __shared__ __align__(16) unsigned short sH[TILE * OO];      //  8 KB bf16
  __shared__ float sBias[4 * OO];
  __shared__ float sConvB[OO];

  const int l        = blockIdx.y;
  const int tileBase = blockIdx.x * TILE;
  const int tid  = threadIdx.x;
  const int wave = tid >> 5;
  const int lane = tid & 31;
  const int m0   = wave << 4;     // this wave owns rows m0..m0+15
  const int n16  = lane & 15;
  const int hi   = lane >> 4;

  // Stage weights as bf16 in LDS; fold the two biases.
  for (int i = tid; i < 4 * OO * OO; i += NTHREADS) {
    sWih[i] = f2bf(W_ih[(size_t)l * 4 * OO * OO + i]);
    sWhh[i] = f2bf(W_hh[(size_t)l * 4 * OO * OO + i]);
  }
  for (int i = tid; i < 4 * OO; i += NTHREADS)
    sBias[i] = b_ih[l * 4 * OO + i] + b_hh[l * 4 * OO + i];
  for (int i = tid; i < OO; i += NTHREADS) sConvB[i] = conv_b[l * OO + i];
  for (int i = tid; i < TILE * OO; i += NTHREADS) sH[i] = 0;  // h0 = 0
  __syncthreads();

  float creg[4][8];   // c state kept in WMMA D-layout registers
#pragma unroll
  for (int jg = 0; jg < 4; ++jg)
#pragma unroll
    for (int r = 0; r < 8; ++r) creg[jg][r] = 0.0f;

  for (int t = 0; t < TT; ++t) {
    const float* Pt = P + ((size_t)(l * TT + t)) * NN * OO;

    // ---- gather-max over K neighbors, ego-subtract, + conv bias -> bf16 LDS
    for (int r = 0; r < 16; ++r) {
      const int m    = m0 + r;
      const int node = tileBase + m;
      float ax, ay;
      if (node < NN) {                       // wave-uniform branch
        ax = -3.402823466e38f; ay = -3.402823466e38f;
        const int* arow = As + ((size_t)(l * TT + t) * NN + node) * KK;
#pragma unroll
        for (int k = 0; k < KK; ++k) {
          const int idx = arow[k];           // uniform -> scalar load
          const float2 v = *(const float2*)(Pt + (size_t)idx * OO + 2 * lane);
          ax = fmaxf(ax, v.x); ay = fmaxf(ay, v.y);
        }
        const float2 e = *(const float2*)(Pt + (size_t)node * OO + 2 * lane);
        ax = ax - e.x + sConvB[2 * lane];
        ay = ay - e.y + sConvB[2 * lane + 1];
      } else { ax = 0.0f; ay = 0.0f; }
      const unsigned int pk =
          (unsigned int)f2bf(ax) | ((unsigned int)f2bf(ay) << 16);
      *(unsigned int*)&sFeat[m * OO + 2 * lane] = pk;
    }
    // No barrier needed: each wave exclusively owns its 16 rows of
    // sFeat/sH; weights are read-only after the initial barrier.

    // ---- gates = feat@Wih^T + h@Whh^T + b via v_wmma_f32_16x16x32_bf16 ----
    const v16bf fa0 = load_afrag(sFeat, m0, 0, lane);
    const v16bf fa1 = load_afrag(sFeat, m0, 32, lane);
    const v16bf ha0 = load_afrag(sH, m0, 0, lane);
    const v16bf ha1 = load_afrag(sH, m0, 32, lane);

#pragma unroll
    for (int jg = 0; jg < 4; ++jg) {
      const int ji = jg * 16, jf = 64 + jg * 16,
                jc = 128 + jg * 16, jo = 192 + jg * 16;
      v8f aI = {0.f,0.f,0.f,0.f,0.f,0.f,0.f,0.f};
      v8f aF = aI, aG = aI, aO = aI;

      aI = wmma_bf16(fa0, load_bfrag(sWih, ji, 0, lane), aI);
      aI = wmma_bf16(fa1, load_bfrag(sWih, ji, 32, lane), aI);
      aI = wmma_bf16(ha0, load_bfrag(sWhh, ji, 0, lane), aI);
      aI = wmma_bf16(ha1, load_bfrag(sWhh, ji, 32, lane), aI);

      aF = wmma_bf16(fa0, load_bfrag(sWih, jf, 0, lane), aF);
      aF = wmma_bf16(fa1, load_bfrag(sWih, jf, 32, lane), aF);
      aF = wmma_bf16(ha0, load_bfrag(sWhh, jf, 0, lane), aF);
      aF = wmma_bf16(ha1, load_bfrag(sWhh, jf, 32, lane), aF);

      aG = wmma_bf16(fa0, load_bfrag(sWih, jc, 0, lane), aG);
      aG = wmma_bf16(fa1, load_bfrag(sWih, jc, 32, lane), aG);
      aG = wmma_bf16(ha0, load_bfrag(sWhh, jc, 0, lane), aG);
      aG = wmma_bf16(ha1, load_bfrag(sWhh, jc, 32, lane), aG);

      aO = wmma_bf16(fa0, load_bfrag(sWih, jo, 0, lane), aO);
      aO = wmma_bf16(fa1, load_bfrag(sWih, jo, 32, lane), aO);
      aO = wmma_bf16(ha0, load_bfrag(sWhh, jo, 0, lane), aO);
      aO = wmma_bf16(ha1, load_bfrag(sWhh, jo, 32, lane), aO);

      const float bI = sBias[ji + n16], bF = sBias[jf + n16];
      const float bG = sBias[jc + n16], bO = sBias[jo + n16];

#pragma unroll
      for (int r = 0; r < 8; ++r) {
        const int m = m0 + r + (hi << 3);   // D-layout row
        const float gi = 1.0f / (1.0f + __expf(-(aI[r] + bI)));
        const float gf = 1.0f / (1.0f + __expf(-(aF[r] + bF)));
        const float gg = tanhf(aG[r] + bG);
        const float go = 1.0f / (1.0f + __expf(-(aO[r] + bO)));
        const float cv = gf * creg[jg][r] + gi * gg;
        creg[jg][r] = cv;
        const float hv = go * tanhf(cv);
        sH[m * OO + jg * 16 + n16] = f2bf(hv);   // re-layout for next A-frag
        const int node = tileBase + m;
        if (node < NN) {
          // max over levels, accumulated with atomics (out pre-set to -inf)
          atomicMaxF(out + ((size_t)node * TT + t) * OO + jg * 16 + n16, hv);
          if (t == TT - 1) {
            atomicMaxF(out + (size_t)NN * TT * OO +
                       (size_t)node * OO + jg * 16 + n16, hv);
            atomicMaxF(out + (size_t)NN * TT * OO + (size_t)NN * OO +
                       (size_t)node * OO + jg * 16 + n16, cv);
          }
        }
      }
    }
  }
}

extern "C" void kernel_launch(void* const* d_in, const int* in_sizes, int n_in,
                              void* d_out, int out_size, void* d_ws,
                              size_t ws_size, hipStream_t stream) {
  const float* X      = (const float*)d_in[0];
  const int*   As     = (const int*)  d_in[1];
  // d_in[2] key_out, d_in[3] key_states: unused by reference path
  const float* conv_w = (const float*)d_in[4];
  const float* conv_b = (const float*)d_in[5];
  const float* W_ih   = (const float*)d_in[6];
  const float* W_hh   = (const float*)d_in[7];
  const float* b_ih   = (const float*)d_in[8];
  const float* b_hh   = (const float*)d_in[9];
  float* out = (float*)d_out;
  float* P   = (float*)d_ws;   // [L,T,N,O] f32 = 61.44 MB scratch

  const int total_out = NN * TT * OO + 2 * NN * OO;  // 6,400,000
  init_out_kernel<<<(total_out + 255) / 256, 256, 0, stream>>>(out, total_out);

  const int ntiles = (NN + 63) / 64;  // 157
  conv_p_kernel<<<dim3(ntiles, TT, LL), 256, 0, stream>>>(X, conv_w, P);

  dgcn_lstm_kernel<<<dim3((NN + TILE - 1) / TILE, LL), NTHREADS, 0, stream>>>(
      P, As, conv_b, W_ih, W_hh, b_ih, b_hh, out);
}
